// LSTMSeq2Seq_43164421325010
// MI455X (gfx1250) — compile-verified, matrix operand order checked by reading, not run on hardware
//
#include <hip/hip_runtime.h>

// ---------------------------------------------------------------------------
// LSTM seq2seq for MI455X (gfx1250, wave32, WMMA).
// v4: wave = 32 batch rows x 16 hidden cols x ALL 4 gates (8 accumulators,
// register-level gate fusion -> zero LDS / zero barriers).  2 A-frags +
// 4 B-frags per K-chunk (12 b128 loads per 8 WMMAs), manually double-buffered
// one chunk ahead + speculative prefetch two chunks ahead so v_wmma never
// waits on a fresh L2 round trip.  Weights pre-transposed [N][K] bf16.
// ---------------------------------------------------------------------------

typedef __bf16 bf16_t;
typedef __attribute__((ext_vector_type(16))) __bf16 v16bf;
typedef __attribute__((ext_vector_type(8)))  __bf16 v8bf;
typedef __attribute__((ext_vector_type(8)))  float  v8f;

#define HIDN   512
#define NTOT   2048      // 4*HID
#define BATCHN 1024
#define PASTN  256
#define FUTN   64

__device__ __forceinline__ float sigmoidf_(float x) {
    return 1.0f / (1.0f + __expf(-x));
}

// fragment = 16 bf16 per lane: elements k0+lh*8 .. +7  and  +16 .. +23
// (CDNA5 16-bit A/B striping: lane<16 -> K {0..7,16..23}, lane>=16 -> {8..15,24..31})
__device__ __forceinline__ v16bf load_frag_bf(const bf16_t* p) {
    union { v16bf v; struct { v8bf lo, hi; } h; } u;
    u.h.lo = *(const v8bf*)p;
    u.h.hi = *(const v8bf*)(p + 16);
    return u.v;
}
__device__ __forceinline__ v16bf load_frag_f32(const float* p) {
    union { v16bf v; struct { v8bf lo, hi; } h; } u;
    #pragma unroll
    for (int i = 0; i < 8; ++i) {
        u.h.lo[i] = (bf16_t)p[i];
        u.h.hi[i] = (bf16_t)p[i + 16];
    }
    return u.v;
}

#define WMMA_BF16(A, B, C) \
    __builtin_amdgcn_wmma_f32_16x16x32_bf16(false, (A), false, (B), (short)0, (C), false, false)

struct FragSet {
    v16bf a[2];   // 2 row tiles (32 rows)
    v16bf b[4];   // 4 gates x 1 col tile (16 cols)
};

__device__ __forceinline__ void load_set_bf(FragSet& s,
                                            const bf16_t* ap0, const bf16_t* ap1,
                                            const bf16_t* const (&bp)[4], int k0) {
    s.a[0] = load_frag_bf(ap0 + k0);
    s.a[1] = load_frag_bf(ap1 + k0);
    #pragma unroll
    for (int g = 0; g < 4; ++g) s.b[g] = load_frag_bf(bp[g] + k0);
}
__device__ __forceinline__ void load_set_f32a(FragSet& s,
                                              const float* ap0, const float* ap1,
                                              const bf16_t* const (&bp)[4], int k0) {
    s.a[0] = load_frag_f32(ap0 + k0);
    s.a[1] = load_frag_f32(ap1 + k0);
    #pragma unroll
    for (int g = 0; g < 4; ++g) s.b[g] = load_frag_bf(bp[g] + k0);
}
__device__ __forceinline__ void do_wmma8(v8f (&acc)[4][2], const FragSet& s) {
    #pragma unroll
    for (int g = 0; g < 4; ++g)
        #pragma unroll
        for (int mi = 0; mi < 2; ++mi)
            acc[g][mi] = WMMA_BF16(s.a[mi], s.b[g], acc[g][mi]);
}

// Block: 128 threads = 4 waves; wave w owns rows mb*128 + w*32 .. +31 and all
// 4 gates of hidden cols jb*16 .. +15.  acc[gate][row-tile] : 8 x v8f.
template<bool XBF>
__global__ __launch_bounds__(128, 1)
void lstm_step_kernel(const void* __restrict__ Xv, int ldx, int kx,
                      const bf16_t* __restrict__ WxT,  // [2048][kx]  bf16 (transposed)
                      const bf16_t* __restrict__ Hb,   // [B][512]    bf16
                      const bf16_t* __restrict__ UT,   // [2048][512] bf16 (transposed)
                      const float*  __restrict__ bias, // [2048]
                      const float*  __restrict__ c_in,
                      float*        __restrict__ c_out,
                      float*        __restrict__ h_out_f,  // optional
                      bf16_t*       __restrict__ h_out_b)
{
    const int tid  = threadIdx.x;
    const int lane = tid & 31;
    const int wv   = tid >> 5;      // wave 0..3
    const int jb   = blockIdx.x;    // hidden-col block (16 cols), 0..31
    const int mb   = blockIdx.y;    // batch-row block (128 rows), 0..7
    const int l15  = lane & 15;
    const int lh   = lane >> 4;     // K-half select for fragments

    const int grow0 = mb * 128 + wv * 32;  // wave's 32 rows (tiles at +0 / +16)
    const int jg    = jb * 16 + l15;       // lane's hidden column

    // accumulators pre-loaded with bias (per-column constant across C rows)
    v8f acc[4][2];
    #pragma unroll
    for (int g = 0; g < 4; ++g) {
        const float bv = bias[g * HIDN + jg];
        #pragma unroll
        for (int mi = 0; mi < 2; ++mi)
            #pragma unroll
            for (int q = 0; q < 8; ++q) acc[g][mi][q] = bv;
    }

    #pragma unroll
    for (int phase = 0; phase < 2; ++phase) {
        const int      K  = phase ? HIDN : kx;
        const int      ld = phase ? HIDN : ldx;
        const bf16_t*  bw = phase ? UT   : WxT;
        const bf16_t*  bp[4];
        #pragma unroll
        for (int g = 0; g < 4; ++g)
            bp[g] = bw + (size_t)(g * HIDN + jg) * K + lh * 8;

        if (phase || XBF) {
            const bf16_t* xb  = phase ? Hb : (const bf16_t*)Xv;
            const bf16_t* ap0 = xb + (size_t)(grow0      + l15) * ld + lh * 8;
            const bf16_t* ap1 = xb + (size_t)(grow0 + 16 + l15) * ld + lh * 8;
            FragSet cur, nxt;
            load_set_bf(cur, ap0, ap1, bp, 0);
            int k0 = 0;
            #pragma unroll 2
            for (; k0 + 32 < K; k0 += 32) {
                load_set_bf(nxt, ap0, ap1, bp, k0 + 32);
                #pragma unroll
                for (int g = 0; g < 4; ++g)
                    __builtin_prefetch(bp[g] + k0 + 64, 0, 1);
                do_wmma8(acc, cur);
                cur = nxt;
            }
            do_wmma8(acc, cur);
        } else {
            const float* xf  = (const float*)Xv;
            const float* ap0 = xf + (size_t)(grow0      + l15) * ld + lh * 8;
            const float* ap1 = xf + (size_t)(grow0 + 16 + l15) * ld + lh * 8;
            FragSet cur, nxt;
            load_set_f32a(cur, ap0, ap1, bp, 0);
            int k0 = 0;
            #pragma unroll 2
            for (; k0 + 32 < K; k0 += 32) {
                load_set_f32a(nxt, ap0, ap1, bp, k0 + 32);
                do_wmma8(acc, cur);
                cur = nxt;
            }
            do_wmma8(acc, cur);
        }
    }

    // ---- register-level gate fusion: acc[0..3] = (zi, zf, zg, zo) ---------
    // C layout: lane element v -> row = v + lh*8 (within 16-row tile), col = l15
    #pragma unroll
    for (int mi = 0; mi < 2; ++mi) {
        #pragma unroll
        for (int v = 0; v < 8; ++v) {
            const int grow = grow0 + mi * 16 + lh * 8 + v;
            const float ig = sigmoidf_(acc[0][mi][v]);
            const float fg = sigmoidf_(acc[1][mi][v]);
            const float zg = acc[2][mi][v];
            const float og = sigmoidf_(acc[3][mi][v]);
            const int idx = grow * HIDN + jg;
            const float cn = fg * c_in[idx] + ig * zg;
            const float hn = og * cn;
            c_out[idx] = cn;
            if (h_out_f) h_out_f[idx] = hn;
            h_out_b[idx] = (bf16_t)hn;
        }
    }
}

// y = h @ fc_W + fc_b ; also feeds decoder's next x (padded K=32 slot)
__global__ __launch_bounds__(256)
void fc_kernel(const float* __restrict__ h, const float* __restrict__ fcW,
               const float* __restrict__ fcb, float* __restrict__ out,
               float* __restrict__ xdec, int t)
{
    const int i = blockIdx.x * 256 + threadIdx.x;   // 0..8191
    const int b = i >> 3;
    const int o = i & 7;
    const float* hr = h + (size_t)b * HIDN;
    float acc = fcb[o];
    #pragma unroll 8
    for (int k = 0; k < HIDN; ++k) acc += hr[k] * fcW[k * 8 + o];
    out[(size_t)(b * FUTN + t) * 8 + o] = acc;
    xdec[b * 32 + o] = acc;
}

__global__ void zero_u32_kernel(unsigned int* p, int n) {
    int i = blockIdx.x * 256 + threadIdx.x;
    if (i < n) p[i] = 0u;
}

// W [Ksrc][2048] fp32  ->  WT [2048][Kdst] bf16 (transposed, Kdst >= Ksrc padded)
__global__ void cvt_transpose_kernel(const float* __restrict__ W,
                                     bf16_t* __restrict__ WT,
                                     int Ksrc, int Kdst) {
    int i = blockIdx.x * 256 + threadIdx.x;
    if (i >= Ksrc * NTOT) return;
    int n = i / Ksrc;
    int k = i - n * Ksrc;
    WT[(size_t)n * Kdst + k] = (bf16_t)W[(size_t)k * NTOT + n];
}

__global__ void init_x0_kernel(const float* __restrict__ inseq,
                               float* __restrict__ xdec) {
    int i = blockIdx.x * 256 + threadIdx.x;  // 8192
    int b = i >> 3, o = i & 7;
    xdec[b * 32 + o] = inseq[(size_t)(b * PASTN + (PASTN - 1)) * 64 + o];
}

extern "C" void kernel_launch(void* const* d_in, const int* in_sizes, int n_in,
                              void* d_out, int out_size, void* d_ws, size_t ws_size,
                              hipStream_t stream) {
    (void)in_sizes; (void)n_in; (void)out_size; (void)ws_size;
    const float* input_seq = (const float*)d_in[0];
    const float* enc0_W = (const float*)d_in[1];
    const float* enc0_U = (const float*)d_in[2];
    const float* enc0_b = (const float*)d_in[3];
    const float* enc1_W = (const float*)d_in[4];
    const float* enc1_U = (const float*)d_in[5];
    const float* enc1_b = (const float*)d_in[6];
    const float* dec0_W = (const float*)d_in[7];
    const float* dec0_U = (const float*)d_in[8];
    const float* dec0_b = (const float*)d_in[9];
    const float* dec1_W = (const float*)d_in[10];
    const float* dec1_U = (const float*)d_in[11];
    const float* dec1_b = (const float*)d_in[12];
    const float* fc_W   = (const float*)d_in[13];
    const float* fc_b   = (const float*)d_in[14];
    float* out = (float*)d_out;

    // ---------------- workspace carve-up ----------------
    char* ws = (char*)d_ws;
    size_t off = 0;
    auto carve = [&](size_t bytes) -> char* {
        char* p = ws + off;
        off += (bytes + 255) & ~(size_t)255;
        return p;
    };
    bf16_t* w_e0W = (bf16_t*)carve((size_t)NTOT * 64  * 2);  // [2048][64]
    bf16_t* w_e0U = (bf16_t*)carve((size_t)NTOT * 512 * 2);  // [2048][512]
    bf16_t* w_e1W = (bf16_t*)carve((size_t)NTOT * 512 * 2);
    bf16_t* w_e1U = (bf16_t*)carve((size_t)NTOT * 512 * 2);
    bf16_t* w_d0W = (bf16_t*)carve((size_t)NTOT * 32  * 2);  // [2048][32], 8 valid
    bf16_t* w_d0U = (bf16_t*)carve((size_t)NTOT * 512 * 2);
    bf16_t* w_d1W = (bf16_t*)carve((size_t)NTOT * 512 * 2);
    bf16_t* w_d1U = (bf16_t*)carve((size_t)NTOT * 512 * 2);
    bf16_t* seq0  = (bf16_t*)carve((size_t)PASTN * BATCHN * HIDN * 2); // layer0 outputs
    bf16_t* h1A   = (bf16_t*)carve((size_t)BATCHN * HIDN * 2);
    bf16_t* h1B   = (bf16_t*)carve((size_t)BATCHN * HIDN * 2);
    bf16_t* hzero = (bf16_t*)carve((size_t)BATCHN * HIDN * 2);
    float*  hf    = (float*) carve((size_t)BATCHN * HIDN * 4);
    float*  cbuf  = (float*) carve((size_t)BATCHN * HIDN * 4);
    float*  xdec  = (float*) carve((size_t)BATCHN * 32 * 4);

    #define ZERO_U32(p, n32) zero_u32_kernel<<<((n32) + 255) / 256, 256, 0, stream>>>((unsigned int*)(p), (n32))
    #define TCVT(s, d, ks, kd) cvt_transpose_kernel<<<((ks) * NTOT + 255) / 256, 256, 0, stream>>>((s), (d), (ks), (kd))

    // weights -> transposed bf16 [N][K] (dec0_W zero-padded K 8->32)
    TCVT(enc0_W, w_e0W, 64, 64);
    TCVT(enc0_U, w_e0U, 512, 512);
    TCVT(enc1_W, w_e1W, 512, 512);
    TCVT(enc1_U, w_e1U, 512, 512);
    ZERO_U32(w_d0W, NTOT * 32 * 2 / 4);
    TCVT(dec0_W, w_d0W, 8, 32);
    TCVT(dec0_U, w_d0U, 512, 512);
    TCVT(dec1_W, w_d1W, 512, 512);
    TCVT(dec1_U, w_d1U, 512, 512);
    // state init
    ZERO_U32(cbuf,  BATCHN * HIDN);          // fp32 c = 0
    ZERO_U32(hzero, BATCHN * HIDN * 2 / 4);  // bf16 h = 0
    ZERO_U32(xdec,  BATCHN * 32);            // padded decoder x
    init_x0_kernel<<<32, 256, 0, stream>>>(input_seq, xdec);

    const dim3 grid(32, 8);    // 32 hidden-col(16) blocks x 8 batch-row(128) blocks
    const size_t HSTEP = (size_t)BATCHN * HIDN;

    // ---------------- encoder layer 0 (256 steps, outputs -> seq0) ---------
    for (int t = 0; t < PASTN; ++t) {
        const bf16_t* Hprev = (t == 0) ? hzero : seq0 + (size_t)(t - 1) * HSTEP;
        lstm_step_kernel<false><<<grid, 128, 0, stream>>>(
            input_seq + (size_t)t * 64, PASTN * 64, 64,
            w_e0W, Hprev, w_e0U, enc0_b, cbuf, cbuf, nullptr,
            seq0 + (size_t)t * HSTEP);
    }
    // ---------------- encoder layer 1: initial carry = layer0 FINAL state --
    const bf16_t* cur = seq0 + (size_t)(PASTN - 1) * HSTEP;  // h = h0_final
    bf16_t* nxt = h1A;                                       // c continues in cbuf
    for (int t = 0; t < PASTN; ++t) {
        lstm_step_kernel<true><<<grid, 128, 0, stream>>>(
            seq0 + (size_t)t * HSTEP, HIDN, HIDN,
            w_e1W, cur, w_e1U, enc1_b, cbuf, cbuf, nullptr, nxt);
        cur = nxt;
        nxt = (nxt == h1A) ? h1B : h1A;
    }
    // ---------------- decoder: single (h,c) through both cells -------------
    for (int t = 0; t < FUTN; ++t) {
        // dec0: x (padded K=32, fp32), H = h
        lstm_step_kernel<false><<<grid, 128, 0, stream>>>(
            xdec, 32, 32, w_d0W, cur, w_d0U, dec0_b, cbuf, cbuf, nullptr, nxt);
        cur = nxt; nxt = (nxt == h1A) ? h1B : h1A;
        // dec1: X = H = h   (z = h@W + h@U)
        lstm_step_kernel<true><<<grid, 128, 0, stream>>>(
            cur, HIDN, HIDN, w_d1W, cur, w_d1U, dec1_b, cbuf, cbuf, hf, nxt);
        cur = nxt; nxt = (nxt == h1A) ? h1B : h1A;
        // y = h @ fc_W + fc_b  -> out[:, t, :] and next decoder x
        fc_kernel<<<32, 256, 0, stream>>>(hf, fc_W, fc_b, out, xdec, t);
    }
    #undef ZERO_U32
    #undef TCVT
}
